// MetricPatchEmbed_1589137899652
// MI455X (gfx1250) — matrix operand order repaired and checked by reading, not compiled, not run on hardware
//
#include <hip/hip_runtime.h>

typedef _Float16 v16h __attribute__((ext_vector_type(16)));
typedef _Float16 v8h  __attribute__((ext_vector_type(8)));
typedef float    v8f  __attribute__((ext_vector_type(8)));

// Problem constants (from reference)
constexpr int NB   = 64;              // batch
constexpr int Cc   = 3;               // channels
constexpr int Hh   = 224, Ww = 224;
constexpr int PP   = 7;               // patch size
constexpr int Hp   = 32, Wp = 32;     // patches per side
constexpr int Ee   = 384;             // embed dim
constexpr int Kk   = 147;             // C*P*P
constexpr int Kpad = 160;             // padded K (5 x 32 WMMA steps)
constexpr int Mrows = NB * Hp * Wp;   // 65536 GEMM rows

__device__ __forceinline__ float sigm(float v) { return 1.0f / (1.0f + __expf(-v)); }

// ---------------------------------------------------------------------------
// Kernel 0: pack proj_w (f32 [E][147]) -> f16 B matrix [E][Kpad], zero padded.
// ---------------------------------------------------------------------------
__global__ void pack_b(const float* __restrict__ pw, _Float16* __restrict__ Bm) {
    int idx = blockIdx.x * 256 + threadIdx.x;
    if (idx >= Ee * Kpad) return;
    int e = idx / Kpad, k = idx - e * Kpad;
    Bm[idx] = (k < Kk) ? (_Float16)pw[e * Kk + k] : (_Float16)0.0f;
}

// ---------------------------------------------------------------------------
// Kernel 1: fused metric-conv + deformed bilinear sampler.
// One 64-thread block per patch; writes one f16 row of A ([Mrows][Kpad]).
// Sample point for direction d: (hp*7+3 + s*uy/(F+eps), wp*7+3 + s*ux/(F+eps))
// (the grid terms cancel in the reference arithmetic).
// ---------------------------------------------------------------------------
__global__ __launch_bounds__(64) void sampler(const float* __restrict__ x,
                                              const float* __restrict__ mw,
                                              const float* __restrict__ mb,
                                              _Float16* __restrict__ A) {
    __shared__ float xs[Kk];
    __shared__ float prm[7];
    __shared__ float drv[5];   // m00, m01, m11, w0, w1

    const int patch = blockIdx.x;
    const int wp = patch & 31;
    const int hp = (patch >> 5) & 31;
    const int b  = patch >> 10;
    const int t  = threadIdx.x;

    // stage the 7x7x3 input patch in LDS (k = c*49 + i*7 + j)
    for (int k = t; k < Kk; k += 64) {
        int c = k / 49, r = k - c * 49, i = r / 7, j = r - i * 7;
        xs[k] = x[(((size_t)b * Cc + c) * Hh + hp * PP + i) * Ww + wp * PP + j];
    }
    __syncthreads();

    // 7-channel metric conv (147-long dots)
    if (t < 7) {
        float acc = mb[t];
        const float* wrow = mw + t * Kk;
        #pragma unroll 7
        for (int k = 0; k < Kk; ++k) acc += xs[k] * wrow[k];
        prm[t] = acc;
    }
    __syncthreads();

    if (t == 0) {
        float p0 = prm[0], p1 = prm[1];
        float nrm = fmaxf(sqrtf(p0 * p0 + p1 * p1), 1e-12f);
        float v0 = p0 / nrm, v1 = p1 / nrm;
        float sc = 0.5f + 1.5f * sigm(prm[4]);
        float e0 = 2.0f * sigm(prm[2]) * sc;
        float e1 = 2.0f * sigm(prm[3]) * sc;
        drv[0] = e0 * v0 * v0 + e1 * v1 * v1;        // m00
        drv[1] = (e0 - e1) * v0 * v1;                // m01
        drv[2] = e0 * v1 * v1 + e1 * v0 * v0;        // m11
        float w0 = prm[5], w1 = prm[6];
        float ws = 0.5f * sigm(sqrtf(w0 * w0 + w1 * w1));  // (1-EPS_W)*sigmoid(|w|)
        drv[3] = w0 * ws;
        drv[4] = w1 * ws;
    }
    __syncthreads();

    if (t < 49) {
        // direction t: ring structure 1 + 8 + 16 + 24
        float ux, uy, s;
        if (t == 0)      { ux = 1.0f; uy = 0.0f; s = 0.0f; }
        else {
            float th;
            if (t <= 8)       { th = 6.2831853071795864f * (float)(t - 1) / 8.0f;  s = 1.0f / 3.0f; }
            else if (t <= 24) { th = 6.2831853071795864f * (float)(t - 9) / 16.0f; s = 2.0f / 3.0f; }
            else              { th = 6.2831853071795864f * (float)(t - 25) / 24.0f; s = 1.0f; }
            ux = cosf(th); uy = sinf(th);
        }
        float m00 = drv[0], m01 = drv[1], m11 = drv[2], w0 = drv[3], w1 = drv[4];
        float quad  = ux * (m00 * ux + m01 * uy) + uy * (m01 * ux + m11 * uy);
        float drift = w0 * ux + w1 * uy;
        float F = sqrtf(quad + 1e-6f) + drift;
        float inv = s / (F + 1e-6f);
        float py = (float)(hp * PP + 3) + uy * inv;
        float px = (float)(wp * PP + 3) + ux * inv;

        float y0f = floorf(py), x0f = floorf(px);
        float wy = py - y0f, wx = px - x0f;
        float w00 = (1.0f - wy) * (1.0f - wx);
        float w01 = (1.0f - wy) * wx;
        float w10 = wy * (1.0f - wx);
        float w11 = wy * wx;

        #pragma unroll
        for (int c = 0; c < Cc; ++c) {
            const float* img = x + ((size_t)b * Cc + c) * (size_t)(Hh * Ww);
            float acc = 0.0f;
            #pragma unroll
            for (int q = 0; q < 4; ++q) {
                float yi = y0f + (float)(q >> 1);
                float xi = x0f + (float)(q & 1);
                bool valid = (yi >= 0.0f) && (yi <= (float)(Hh - 1)) &&
                             (xi >= 0.0f) && (xi <= (float)(Ww - 1));
                int yc = (int)fminf(fmaxf(yi, 0.0f), (float)(Hh - 1));
                int xc = (int)fminf(fmaxf(xi, 0.0f), (float)(Ww - 1));
                float v = img[(size_t)yc * Ww + xc];
                float wq = (q == 0) ? w00 : (q == 1) ? w01 : (q == 2) ? w10 : w11;
                acc += (valid ? v : 0.0f) * wq;
            }
            A[(size_t)patch * Kpad + c * 49 + t] = (_Float16)acc;
        }
    } else if (t < 62) {        // zero the K padding 147..159
        A[(size_t)patch * Kpad + Kk + (t - 49)] = (_Float16)0.0f;
    }
}

// ---------------------------------------------------------------------------
// Kernel 2: WMMA GEMM. out[m][n] = sum_k A[m][k] * Bm[n][k] + bias[n]
// M=65536, N=384, K=160. 8 waves / block in 2(M) x 4(N); each wave: 64x16 tile.
// Fragment layouts per CDNA5 ISA:
//   A (16x32 f16): lane m = lane%16; elem e -> K = (e<8 ? e : e+8) + 8*(lane/16)
//   B (32x16 f16): lane n = lane%16; elem e -> K = (e<8 ? e : e+8) + 8*(lane/16)
//   C/D (16x16 f32): n = lane%16; m = vgpr + 8*(lane/16)
// ---------------------------------------------------------------------------
__global__ __launch_bounds__(256) void gemm_wmma(const _Float16* __restrict__ A,
                                                 const _Float16* __restrict__ Bm,
                                                 const float* __restrict__ bias,
                                                 float* __restrict__ out) {
    const int lane = threadIdx.x & 31;
    const int wave = threadIdx.x >> 5;
    const int wm = wave & 1, wn = wave >> 1;
    const int m0 = blockIdx.x * 128 + wm * 64;
    const int n0 = blockIdx.y * 64 + wn * 16;
    const int l16 = lane & 15, hf = lane >> 4;

    v8f acc[4] = {};
    const _Float16* bcol  = Bm + (size_t)(n0 + l16) * Kpad + 8 * hf;
    const _Float16* abase = A + (size_t)(m0 + l16) * Kpad + 8 * hf;

    #pragma unroll
    for (int kb = 0; kb < Kpad; kb += 32) {
        v16h bf = __builtin_shufflevector(*(const v8h*)(bcol + kb), *(const v8h*)(bcol + kb + 16),
                                          0, 1, 2, 3, 4, 5, 6, 7, 8, 9, 10, 11, 12, 13, 14, 15);
        #pragma unroll
        for (int s = 0; s < 4; ++s) {
            const _Float16* ar = abase + (size_t)(s * 16) * Kpad + kb;
            v8h lo = *(const v8h*)(ar);                 // K = kb + 8*hf + 0..7
            v8h hi = *(const v8h*)(ar + 16);            // K = kb + 16 + 8*hf + 0..7
            v16h af = __builtin_shufflevector(lo, hi, 0, 1, 2, 3, 4, 5, 6, 7,
                                                      8, 9, 10, 11, 12, 13, 14, 15);
            acc[s] = __builtin_amdgcn_wmma_f32_16x16x32_f16(
                false, af, false, bf, (short)0, acc[s], false, false);
        }
    }

    float bv = bias[n0 + l16];
    #pragma unroll
    for (int s = 0; s < 4; ++s) {
        #pragma unroll
        for (int r = 0; r < 8; ++r) {
            int m = m0 + s * 16 + hf * 8 + r;
            out[(size_t)m * Ee + n0 + l16] = acc[s][r] + bv;
        }
    }
}

// ---------------------------------------------------------------------------
extern "C" void kernel_launch(void* const* d_in, const int* in_sizes, int n_in,
                              void* d_out, int out_size, void* d_ws, size_t ws_size,
                              hipStream_t stream) {
    (void)in_sizes; (void)n_in; (void)out_size; (void)ws_size;
    const float* x  = (const float*)d_in[0];   // (64,3,224,224)
    const float* mw = (const float*)d_in[1];   // (7,3,7,7)
    const float* mb = (const float*)d_in[2];   // (7,)
    const float* pw = (const float*)d_in[3];   // (384,3,7,7)
    const float* pb = (const float*)d_in[4];   // (384,)
    float* out = (float*)d_out;                // (64,1024,384)

    _Float16* Bm = (_Float16*)d_ws;                         // 384*160*2   = 120 KB
    _Float16* A  = (_Float16*)((char*)d_ws + (128 << 10));  // 65536*160*2 = 20 MB

    pack_b<<<(Ee * Kpad + 255) / 256, 256, 0, stream>>>(pw, Bm);
    sampler<<<Mrows, 64, 0, stream>>>(x, mw, mb, A);
    gemm_wmma<<<dim3(Mrows / 128, Ee / 64), 256, 0, stream>>>(A, Bm, pb, out);
}
